// Qwen3_5Attention_25168508354981
// MI455X (gfx1250) — compile-verified, matrix operand order checked by reading, not run on hardware
//
#include <hip/hip_runtime.h>

#define BB 2
#define SS 2048
#define DMODEL 2048
#define NH 16
#define NKVH 4
#define HD 128
#define NREP (NH / NKVH)

typedef __bf16 bf16;
typedef __attribute__((ext_vector_type(16))) __bf16 v16bf;
typedef __attribute__((ext_vector_type(8)))  __bf16 v8bf;
typedef __attribute__((ext_vector_type(8)))  float  v8f;
typedef int v4i __attribute__((vector_size(16)));

__device__ constexpr float ATTN_SCALE = 0.08838834764831843f; // 1/sqrt(128)

#if defined(__has_builtin)
#if __has_builtin(__builtin_amdgcn_global_load_async_to_lds_b128)
#define HAVE_ASYNC 1
#endif
#endif
#ifndef HAVE_ASYNC
#define HAVE_ASYNC 0
#endif

// 16-byte global -> LDS copy; async (ASYNCcnt-tracked) when available.
// Builtin signature: (V4i addrspace(1)* src, V4i addrspace(3)* dst, Ii off, Ii cpol)
__device__ __forceinline__ void cp16(void* lds, const void* g) {
#if HAVE_ASYNC
  __builtin_amdgcn_global_load_async_to_lds_b128(
      (__attribute__((address_space(1))) v4i*)(void*)(g),
      (__attribute__((address_space(3))) v4i*)(lds), 0, 0);
#else
  *(v8bf*)lds = *(const v8bf*)g;
#endif
}

__device__ __forceinline__ void async_wait_all() {
#if HAVE_ASYNC
#if __has_builtin(__builtin_amdgcn_s_wait_asynccnt)
  __builtin_amdgcn_s_wait_asynccnt(0);
#else
  asm volatile("s_wait_asynccnt 0" ::: "memory");
#endif
#endif
}

__device__ __forceinline__ v8f wmma_bf16(v16bf a, v16bf b, v8f c) {
  return __builtin_amdgcn_wmma_f32_16x16x32_bf16(false, a, false, b, (short)0, c,
                                                 false, false);
}

// Load a 16x32 bf16 fragment (A- or B-style): row (M for A, N for B) = lane%16,
// contraction elements per CDNA5 16-bit WMMA layout:
//   VGPR0-3: k = 8*half + 0..7,  VGPR4-7: k = 16 + 8*half + 0..7
__device__ __forceinline__ v16bf frag_ld(const bf16* base, int stride) {
  const int lane = threadIdx.x & 31;
  const int row  = lane & 15;
  const int half = lane >> 4;
  const bf16* p = base + row * stride;
  v8bf lo = *(const v8bf*)(p + 8 * half);
  v8bf hi = *(const v8bf*)(p + 16 + 8 * half);
  v16bf f;
  f[0]=lo[0];  f[1]=lo[1];  f[2]=lo[2];  f[3]=lo[3];
  f[4]=lo[4];  f[5]=lo[5];  f[6]=lo[6];  f[7]=lo[7];
  f[8]=hi[0];  f[9]=hi[1];  f[10]=hi[2]; f[11]=hi[3];
  f[12]=hi[4]; f[13]=hi[5]; f[14]=hi[6]; f[15]=hi[7];
  return f;
}

// ---------------------------------------------------------------------------
// fp32 -> bf16 conversion (row-major copy)
// ---------------------------------------------------------------------------
__global__ void cvt_bf16_kernel(const float* __restrict__ x, bf16* __restrict__ y,
                                size_t n) {
  size_t i = (size_t)blockIdx.x * 256 + threadIdx.x;
  if (i < n) y[i] = (bf16)x[i];
}

// fp32 W[K,N] -> bf16 W^T[N,K]  (so GEMM B-tiles are pure byte copies)
__global__ void cvt_t_kernel(const float* __restrict__ x, bf16* __restrict__ y,
                             int K, int N) {
  size_t i = (size_t)blockIdx.x * 256 + threadIdx.x; // dest index over N*K
  size_t total = (size_t)K * N;
  if (i >= total) return;
  int k = (int)(i % K);
  size_t n = i / K;
  y[i] = (bf16)x[(size_t)k * N + n];
}

// ---------------------------------------------------------------------------
// C[M,N] (f32) = A[M,K] (bf16 row major) @ B (given as B^T[N,K] bf16 row major)
// workgroup tile 128x128, 8 waves in 4(M)x2(N), wave tile 32x64 (8 accums).
// Double-buffered LDS staged with async global->LDS copies.
// ---------------------------------------------------------------------------
__global__ __launch_bounds__(256) void gemm_bf16_kernel(
    const bf16* __restrict__ A, const bf16* __restrict__ Bt,
    float* __restrict__ C, int N, int K) {
  __shared__ bf16 As[2][128][32];   // [buf][m][k]
  __shared__ bf16 Bs[2][128][32];   // [buf][n][k]
  const int tid  = threadIdx.x;
  const int wid  = tid >> 5;
  const int lane = tid & 31;
  const int r    = lane & 15;
  const int half = lane >> 4;
  const int wm   = (wid & 3) * 32;   // wave M offset within tile
  const int wn   = (wid >> 2) * 64;  // wave N offset within tile
  const size_t gm0 = (size_t)blockIdx.y * 128;
  const size_t gn0 = (size_t)blockIdx.x * 128;

  const v8f zero = {0.f,0.f,0.f,0.f,0.f,0.f,0.f,0.f};
  v8f acc[2][4];
#pragma unroll
  for (int mi = 0; mi < 2; ++mi)
#pragma unroll
    for (int nt = 0; nt < 4; ++nt) acc[mi][nt] = zero;

  const int nk = K >> 5;
  auto stage = [&](int kt, int buf) {
    int k0 = kt << 5;
#pragma unroll
    for (int i = 0; i < 2; ++i) {
      int slot = tid + i * 256;            // 512 slots of 16B each per matrix
      int row = slot >> 2, kc = (slot & 3) * 8;
      cp16(&As[buf][row][kc], &A[(gm0 + row) * (size_t)K + k0 + kc]);
      cp16(&Bs[buf][row][kc], &Bt[(gn0 + row) * (size_t)K + k0 + kc]);
    }
  };

  stage(0, 0);
  async_wait_all();
  __syncthreads();

  for (int kt = 0; kt < nk; ++kt) {
    const int cur = kt & 1;
    if (kt + 1 < nk) stage(kt + 1, cur ^ 1);   // prefetch next tile (async)

    v16bf af0 = frag_ld(&As[cur][wm][0], 32);
    v16bf af1 = frag_ld(&As[cur][wm + 16][0], 32);
#pragma unroll
    for (int nt = 0; nt < 4; ++nt) {
      v16bf bfr = frag_ld(&Bs[cur][wn + nt * 16][0], 32);
      acc[0][nt] = wmma_bf16(af0, bfr, acc[0][nt]);
      acc[1][nt] = wmma_bf16(af1, bfr, acc[1][nt]);
    }
    async_wait_all();
    __syncthreads();
  }

  // C layout: acc element e -> row (e + 8*half), col lane%16
#pragma unroll
  for (int mi = 0; mi < 2; ++mi)
#pragma unroll
    for (int nt = 0; nt < 4; ++nt)
#pragma unroll
      for (int e = 0; e < 8; ++e)
        C[(gm0 + wm + mi * 16 + e + 8 * half) * (size_t)N +
          gn0 + wn + nt * 16 + r] = acc[mi][nt][e];
}

// ---------------------------------------------------------------------------
// RMSNorm + RoPE (first 64 dims), one wave per (b, s, head) row of 128
// dst layout: (B, nh, S, 128) bf16
// ---------------------------------------------------------------------------
__global__ __launch_bounds__(256) void norm_rope_kernel(
    const float* __restrict__ raw, const float* __restrict__ cosb,
    const float* __restrict__ sinb, const float* __restrict__ w,
    bf16* __restrict__ dst, int nh, int rowStride, int headStride, size_t nrows) {
  const int lane = threadIdx.x & 31;
  size_t rowid = (size_t)blockIdx.x * 8 + (threadIdx.x >> 5);
  if (rowid >= nrows) return;
  int h = (int)(rowid % nh);
  size_t bs = rowid / nh;   // b*S + s
  const float* rp = raw + bs * (size_t)rowStride + (size_t)h * headStride;
  float x0 = rp[lane], x1 = rp[lane + 32], x2 = rp[lane + 64], x3 = rp[lane + 96];
  float ssq = x0 * x0 + x1 * x1 + x2 * x2 + x3 * x3;
#pragma unroll
  for (int off = 1; off < 32; off <<= 1) ssq += __shfl_xor(ssq, off, 32);
  float rms = rsqrtf(ssq * (1.0f / 128.0f) + 1e-6f);
  x0 *= rms * w[lane];      x1 *= rms * w[lane + 32];
  x2 *= rms * w[lane + 64]; x3 *= rms * w[lane + 96];
  const float* cp = cosb + bs * 64;
  const float* sp = sinb + bs * 64;
  float c0 = cp[lane], sn0 = sp[lane];
  float c1 = cp[lane + 32], sn1 = sp[lane + 32];
  float y0 = x0 * c0 - x1 * sn0;   // rotate_half pairing
  float y1 = x1 * c1 + x0 * sn1;
  size_t s = bs % SS, b = bs / SS;
  bf16* dp = dst + (((size_t)b * nh + h) * SS + s) * HD;
  dp[lane] = (bf16)y0;        dp[lane + 32] = (bf16)y1;
  dp[lane + 64] = (bf16)x2;   dp[lane + 96] = (bf16)x3;
}

// ---------------------------------------------------------------------------
// V: (B*S, KVH*128) f32 -> V^T (B, KVH, 128, S) bf16  (d-major for byte-copy
// staging of [d][kv] tiles in the attention kernel)
// ---------------------------------------------------------------------------
__global__ void v_pack_kernel(const float* __restrict__ vraw, bf16* __restrict__ vt) {
  size_t i = (size_t)blockIdx.x * 256 + threadIdx.x;  // dest idx (B,KVH,HD,S)
  int s = (int)(i % SS); size_t t = i / SS;
  int d = (int)(t % HD); t /= HD;
  int kv = (int)(t % NKVH);
  int b  = (int)(t / NKVH);
  vt[i] = (bf16)vraw[(((size_t)b * SS + s) * NKVH + kv) * HD + d];
}

// ---------------------------------------------------------------------------
// Flash attention with fused sigmoid gating.
// grid = (S/128, B*NH), 256 threads (8 waves), wave owns 16 q rows.
// Q,K: (B, heads, S, 128) bf16; V: (B, KVH, 128, S) bf16 (pre-transposed).
// Output Og: (B, S, NH*128) bf16, gated.
// ---------------------------------------------------------------------------
__global__ __launch_bounds__(256) void attn_kernel(
    const bf16* __restrict__ Q, const bf16* __restrict__ K,
    const bf16* __restrict__ V, const float* __restrict__ qraw,
    bf16* __restrict__ Og) {
  __shared__ bf16 Ks[32][128];     // [kv_local][d]
  __shared__ bf16 Vs[128][32];     // [d][kv_local]
  __shared__ bf16 Ps[8][16][32];   // per-wave P staging (C-layout -> A-layout)
  const int tid  = threadIdx.x;
  const int wid  = tid >> 5;
  const int lane = tid & 31;
  const int r    = lane & 15;
  const int half = lane >> 4;
  const int bh = blockIdx.y;
  const int b = bh / NH, h = bh % NH;
  const int kvh = h / NREP;
  const int qb = blockIdx.x * 128;

  const bf16* Qp = Q + (((size_t)b * NH + h) * SS + qb) * HD;
  const bf16* Kp = K + (((size_t)b * NKVH + kvh) * SS) * HD;
  const bf16* Vp = V + ((size_t)b * NKVH + kvh) * (size_t)HD * SS;

  // Q fragments: 4 chunks of K=32 held in VGPRs for the whole kernel
  v16bf qf[4];
#pragma unroll
  for (int c = 0; c < 4; ++c)
    qf[c] = frag_ld(Qp + (size_t)(wid * 16) * HD + c * 32, HD);

  const v8f zero = {0.f,0.f,0.f,0.f,0.f,0.f,0.f,0.f};
  v8f oacc[8];
#pragma unroll
  for (int dt = 0; dt < 8; ++dt) oacc[dt] = zero;
  float mrow[8], lrow[8];
#pragma unroll
  for (int e = 0; e < 8; ++e) { mrow[e] = -1e30f; lrow[e] = 0.f; }

  const int q_end = qb + wid * 16 + 15;
  const int nblk = (qb + 128) >> 5;   // causal upper bound for this workgroup

  for (int jb = 0; jb < nblk; ++jb) {
    const int kv0 = jb << 5;
    // async stage of K [kv][d] and V^T [d][kv] blocks (pure byte copies)
#pragma unroll
    for (int i = 0; i < 2; ++i) {
      int slot = tid + i * 256;
      int krow = slot >> 4, d8 = (slot & 15) * 8;
      cp16(&Ks[krow][d8], &Kp[(size_t)(kv0 + krow) * HD + d8]);
      int dr = slot >> 2, c8 = (slot & 3) * 8;
      cp16(&Vs[dr][c8], &Vp[(size_t)dr * SS + kv0 + c8]);
    }
    async_wait_all();
    __syncthreads();

    if (kv0 <= q_end) {   // wave-uniform causal skip
      // S = Q @ K^T over d (two 16-col score tiles)
      v8f s0 = zero, s1 = zero;
#pragma unroll
      for (int c = 0; c < 4; ++c) {
        v16bf kf0 = frag_ld(&Ks[0][c * 32], 128);
        v16bf kf1 = frag_ld(&Ks[16][c * 32], 128);
        s0 = wmma_bf16(qf[c], kf0, s0);
        s1 = wmma_bf16(qf[c], kf1, s1);
      }
      const int col0 = kv0 + r, col1 = kv0 + 16 + r;
#pragma unroll
      for (int e = 0; e < 8; ++e) {
        int rowa = qb + wid * 16 + e + 8 * half;
        float a0 = s0[e] * ATTN_SCALE + ((col0 <= rowa) ? 0.f : -1e9f);
        float a1 = s1[e] * ATTN_SCALE + ((col1 <= rowa) ? 0.f : -1e9f);
        float mx = fmaxf(a0, a1);
#pragma unroll
        for (int off = 1; off < 16; off <<= 1)
          mx = fmaxf(mx, __shfl_xor(mx, off, 32));   // within 16-lane half
        float mnew  = fmaxf(mrow[e], mx);
        float alpha = __expf(mrow[e] - mnew);
        float p0 = __expf(a0 - mnew);
        float p1 = __expf(a1 - mnew);
        float psum = p0 + p1;
#pragma unroll
        for (int off = 1; off < 16; off <<= 1)
          psum += __shfl_xor(psum, off, 32);
        lrow[e] = lrow[e] * alpha + psum;
        mrow[e] = mnew;
#pragma unroll
        for (int dt = 0; dt < 8; ++dt) oacc[dt][e] *= alpha;
        // stage P (C-layout) into per-wave LDS scratch
        Ps[wid][e + 8 * half][r]      = (bf16)p0;
        Ps[wid][e + 8 * half][16 + r] = (bf16)p1;
      }
      // O += P @ V  (A-frag read back from same-wave LDS; DS is in-order)
      v16bf pf = frag_ld(&Ps[wid][0][0], 32);
#pragma unroll
      for (int dt = 0; dt < 8; ++dt) {
        v16bf vf = frag_ld(&Vs[dt * 16][0], 32);
        oacc[dt] = wmma_bf16(pf, vf, oacc[dt]);
      }
    }
    __syncthreads();
  }

  // epilogue: normalize, apply sigmoid(gate), write bf16 (B, S, NH*128)
#pragma unroll
  for (int e = 0; e < 8; ++e) {
    float inv = 1.0f / fmaxf(lrow[e], 1e-20f);
    size_t srow = (size_t)qb + wid * 16 + e + 8 * half;
    const float* gp = qraw + ((size_t)b * SS + srow) * (NH * 2 * HD) +
                      (size_t)h * (2 * HD) + HD;
#pragma unroll
    for (int dt = 0; dt < 8; ++dt) {
      int d = dt * 16 + r;
      float g = gp[d];
      float sig = 1.0f / (1.0f + __expf(-g));
      Og[(((size_t)b * SS + srow) * NH + h) * HD + d] = (bf16)(oacc[dt][e] * inv * sig);
    }
  }
}

// ---------------------------------------------------------------------------
extern "C" void kernel_launch(void* const* d_in, const int* in_sizes, int n_in,
                              void* d_out, int out_size, void* d_ws, size_t ws_size,
                              hipStream_t stream) {
  (void)in_sizes; (void)n_in; (void)out_size; (void)ws_size;
  const float* hs   = (const float*)d_in[0];
  const float* cosb = (const float*)d_in[1];
  const float* sinb = (const float*)d_in[2];
  // d_in[3] = attn_mask (pure causal; generated inline in the attention kernel)
  const float* Wq   = (const float*)d_in[4];
  const float* Wk   = (const float*)d_in[5];
  const float* Wv   = (const float*)d_in[6];
  const float* Wo   = (const float*)d_in[7];
  const float* qw   = (const float*)d_in[8];
  const float* kw   = (const float*)d_in[9];

  char* wptr = (char*)d_ws;
  auto alloc = [&](size_t bytes) -> void* {
    void* p = (void*)wptr;
    wptr += (bytes + 255) & ~(size_t)255;
    return p;
  };
  const size_t M = (size_t)BB * SS;                       // 4096 token rows
  bf16*  hsb  = (bf16*)alloc(M * DMODEL * 2);
  bf16*  wqb  = (bf16*)alloc((size_t)DMODEL * (NH * 2 * HD) * 2);  // W^T
  bf16*  wkb  = (bf16*)alloc((size_t)DMODEL * (NKVH * HD) * 2);    // W^T
  bf16*  wvb  = (bf16*)alloc((size_t)DMODEL * (NKVH * HD) * 2);    // W^T
  bf16*  wob  = (bf16*)alloc((size_t)(NH * HD) * DMODEL * 2);      // W^T
  float* qraw = (float*)alloc(M * (NH * 2 * HD) * 4);
  float* kraw = (float*)alloc(M * (NKVH * HD) * 4);
  float* vraw = (float*)alloc(M * (NKVH * HD) * 4);
  bf16*  qbuf = (bf16*)alloc(M * (NH * HD) * 2);
  bf16*  kbuf = (bf16*)alloc(M * (NKVH * HD) * 2);
  bf16*  vbuf = (bf16*)alloc(M * (NKVH * HD) * 2);   // transposed (B,KVH,HD,S)
  bf16*  og   = (bf16*)alloc(M * (NH * HD) * 2);

  // bf16 conversions: activations row-major, weights transposed to [N][K]
  cvt_bf16_kernel<<<(unsigned)((M * DMODEL + 255) / 256), 256, 0, stream>>>(
      hs, hsb, M * DMODEL);
  {
    size_t n;
    n = (size_t)DMODEL * (NH * 2 * HD);
    cvt_t_kernel<<<(unsigned)((n + 255) / 256), 256, 0, stream>>>(Wq, wqb, DMODEL,
                                                                  NH * 2 * HD);
    n = (size_t)DMODEL * (NKVH * HD);
    cvt_t_kernel<<<(unsigned)((n + 255) / 256), 256, 0, stream>>>(Wk, wkb, DMODEL,
                                                                  NKVH * HD);
    cvt_t_kernel<<<(unsigned)((n + 255) / 256), 256, 0, stream>>>(Wv, wvb, DMODEL,
                                                                  NKVH * HD);
    n = (size_t)(NH * HD) * DMODEL;
    cvt_t_kernel<<<(unsigned)((n + 255) / 256), 256, 0, stream>>>(Wo, wob, NH * HD,
                                                                  DMODEL);
  }

  // QKV projections (N/128 x M/128 workgroups)
  gemm_bf16_kernel<<<dim3((NH * 2 * HD) / 128, (unsigned)(M / 128)), 256, 0, stream>>>(
      hsb, wqb, qraw, NH * 2 * HD, DMODEL);
  gemm_bf16_kernel<<<dim3((NKVH * HD) / 128, (unsigned)(M / 128)), 256, 0, stream>>>(
      hsb, wkb, kraw, NKVH * HD, DMODEL);
  gemm_bf16_kernel<<<dim3((NKVH * HD) / 128, (unsigned)(M / 128)), 256, 0, stream>>>(
      hsb, wvb, vraw, NKVH * HD, DMODEL);

  // RMSNorm + RoPE, layout to (B, heads, S, 128) bf16
  norm_rope_kernel<<<(unsigned)(M * NH / 8), 256, 0, stream>>>(
      qraw, cosb, sinb, qw, qbuf, NH, NH * 2 * HD, 2 * HD, M * NH);
  norm_rope_kernel<<<(unsigned)(M * NKVH / 8), 256, 0, stream>>>(
      kraw, cosb, sinb, kw, kbuf, NKVH, NKVH * HD, HD, M * NKVH);
  v_pack_kernel<<<(unsigned)(M * NKVH * HD / 256), 256, 0, stream>>>(vraw, vbuf);

  // Flash attention + fused sigmoid gating
  attn_kernel<<<dim3(SS / 128, BB * NH), 256, 0, stream>>>(qbuf, kbuf, vbuf, qraw, og);

  // Output projection -> f32 result
  gemm_bf16_kernel<<<dim3(DMODEL / 128, (unsigned)(M / 128)), 256, 0, stream>>>(
      og, wob, (float*)d_out, DMODEL, NH * HD);
}
